// SelfAttention_34737695490204
// MI455X (gfx1250) — matrix operand order repaired and checked by reading, not compiled, not run on hardware
//
#include <hip/hip_runtime.h>
#include <hip/hip_bf16.h>

typedef __attribute__((ext_vector_type(16))) __bf16 v16bf;
typedef __attribute__((ext_vector_type(8)))  __bf16 v8bf;
typedef __attribute__((ext_vector_type(8)))  float  v8f;
typedef __attribute__((ext_vector_type(4))) unsigned int u32x4;
typedef __attribute__((ext_vector_type(8))) int i32x8;
typedef __attribute__((ext_vector_type(4))) int i32x4;

#define B_  4
#define S_  2048
#define F_  512
#define H_  8
#define DH_ 64
#define OUT_ 512
#define N_  (B_ * S_)          // 8192 tokens
#define COLS_ 512              // H*DH
#define NEG_SLOPE 0.01f

#if defined(__has_builtin)
#if __has_builtin(__builtin_amdgcn_tensor_load_to_lds)
#define HAVE_TDM 1
#endif
#endif

// ---------------------------------------------------------------------------
// WMMA helpers
// ---------------------------------------------------------------------------
__device__ __forceinline__ v8f wmma_bf16(v16bf a, v16bf b, v8f c) {
    return __builtin_amdgcn_wmma_f32_16x16x32_bf16(
        false, a, false, b, (short)0, c, false, false);
}

__device__ __forceinline__ v8f zero8() {
    v8f z;
#pragma unroll
    for (int e = 0; e < 8; ++e) z[e] = 0.0f;
    return z;
}

// A fragment (16x32 bf16, row-major src, stride ld). contiguous k order (matches load_bt):
// lane l (g=l>>4, row=l&15): e=0..15 -> k=kBase+16g+e
__device__ __forceinline__ v16bf load_a_bf16(const __bf16* src, int ld,
                                             int row, int kBase, int g) {
    const __bf16* p0 = src + (size_t)row * ld + kBase + 16 * g;
    v8bf lo = *(const v8bf*)(p0);
    v8bf hi = *(const v8bf*)(p0 + 8);
    v16bf a;
#pragma unroll
    for (int e = 0; e < 8; ++e) { a[e] = lo[e]; a[e + 8] = hi[e]; }
    return a;
}

// A fragment from f32 source, converting to bf16 on the fly.
__device__ __forceinline__ v16bf load_a_f32(const float* src, int ld,
                                            int row, int kBase, int g) {
    const float* p0 = src + (size_t)row * ld + kBase + 16 * g;
    v16bf a;
#pragma unroll
    for (int e = 0; e < 8; ++e) {
        a[e]     = (__bf16)p0[e];
        a[e + 8] = (__bf16)p0[e + 8];
    }
    return a;
}

// B fragment (32x16) from transposed storage bt[col][k] (stride ld):
// lane holds column col; elements e -> k = kBase+16g+e (one 32B read).
__device__ __forceinline__ v16bf load_bt(const __bf16* bt, int ld,
                                         int col, int kBase, int g) {
    return *(const v16bf*)(bt + (size_t)col * ld + kBase + 16 * g);
}

// ---------------------------------------------------------------------------
// Tensor Data Mover: async 2D tile (rows x rowElems bf16, row stride
// strideElems) from global -> LDS, laid out contiguously [row][rowElems].
// D# per ISA ch.8: group0 = {count|flags, lds_addr, global_addr, type=2},
// group1 = {mask/data_size, tensor dims, tile dims, dim0 stride}.
// ---------------------------------------------------------------------------
__device__ __forceinline__ void tdm_load_2d(const void* gptr, void* ldsPtr,
                                            unsigned rowElems, unsigned rows,
                                            unsigned long long strideElems) {
#ifdef HAVE_TDM
    unsigned long long ga = (unsigned long long)(size_t)gptr;
    unsigned int la = (unsigned int)(size_t)ldsPtr;   // low 32 bits = LDS byte addr
    u32x4 g0;
    g0[0] = 1u;                                               // count=1, user mode
    g0[1] = la;                                               // lds_addr (bytes)
    g0[2] = (unsigned)(ga & 0xffffffffu);                     // global_addr[31:0]
    g0[3] = (unsigned)((ga >> 32) & 0x01ffffffu) | (2u << 30);// [56:32] | type=2
    const unsigned td0 = 1u << 30, td1 = 1u << 20;            // generous OOB dims
    i32x8 g1;
    g1[0] = (int)(1u << 16);                                  // data_size=1 (2B)
    g1[1] = (int)((td0 & 0xffffu) << 16);                     // tensor_dim0 lo16
    g1[2] = (int)(((td0 >> 16) & 0xffffu) | ((td1 & 0xffffu) << 16));
    g1[3] = (int)(((td1 >> 16) & 0xffffu) | ((rowElems & 0xffffu) << 16)); // tile_dim0
    g1[4] = (int)(rows & 0xffffu);                            // tile_dim1; tile_dim2=0
    g1[5] = (int)(strideElems & 0xffffffffu);                 // dim0_stride lo32
    g1[6] = (int)((strideElems >> 32) & 0xffffu);             // dim0_stride hi16
    g1[7] = 0;
    i32x4 gz; gz[0] = gz[1] = gz[2] = gz[3] = 0;
#if __clang_major__ >= 23
    i32x8 gz8;
#pragma unroll
    for (int e = 0; e < 8; ++e) gz8[e] = 0;
    __builtin_amdgcn_tensor_load_to_lds(g0, g1, gz, gz, gz8, 0);
#else
    __builtin_amdgcn_tensor_load_to_lds(g0, g1, gz, gz, 0);
#endif
#else
    (void)gptr; (void)ldsPtr; (void)rowElems; (void)rows; (void)strideElems;
#endif
}

// Stage a shared B-panel: TDM from wave 0, or cooperative copy fallback.
__device__ __forceinline__ void stage_tile(const __bf16* g, __bf16* lds,
                                           unsigned rowElems, unsigned rows,
                                           unsigned strideElems,
                                           int wave, int tid) {
#ifdef HAVE_TDM
    if (wave == 0) tdm_load_2d(g, lds, rowElems, rows, strideElems);
#else
    int total = (int)(rows * rowElems);
    for (int idx = tid * 16; idx < total; idx += 128 * 16) {
        int r = idx / (int)rowElems, c = idx % (int)rowElems;
        *(v16bf*)(lds + r * rowElems + c) =
            *(const v16bf*)(g + (size_t)r * strideElems + c);
    }
#endif
    (void)wave; (void)tid;
}

__device__ __forceinline__ void finish_stage(bool more_inflight, int wave) {
#ifdef HAVE_TDM
    if (wave == 0) {
#if __has_builtin(__builtin_amdgcn_s_wait_tensorcnt)
        if (more_inflight) __builtin_amdgcn_s_wait_tensorcnt(1);
        else               __builtin_amdgcn_s_wait_tensorcnt(0);
#else
        if (more_inflight) asm volatile("s_wait_tensorcnt 0x1" ::: "memory");
        else               asm volatile("s_wait_tensorcnt 0x0" ::: "memory");
#endif
    }
#endif
    (void)more_inflight; (void)wave;
}

// ---------------------------------------------------------------------------
// Kernel 1: transpose + f32->bf16 convert the four 512x512 weight matrices.
// ---------------------------------------------------------------------------
__global__ void k_transpose_w(const float* s0, const float* s1,
                              const float* s2, const float* s3,
                              __bf16* d0, __bf16* d1, __bf16* d2, __bf16* d3) {
    int z = blockIdx.z;
    const float* s = (z == 0) ? s0 : (z == 1) ? s1 : (z == 2) ? s2 : s3;
    __bf16*      d = (z == 0) ? d0 : (z == 1) ? d1 : (z == 2) ? d2 : d3;
    int idx = blockIdx.x * 256 + threadIdx.x;
    int kk  = idx & 511;
    int n   = idx >> 9;
    d[(size_t)n * 512 + kk] = (__bf16)s[(size_t)kk * 512 + n];
}

// ---------------------------------------------------------------------------
// Kernel 2: QKV projection GEMM, TDM-staged weight panels.
// blockDim=128 (4 waves); wave computes 16x64; all waves share the B panel.
// grid = (512/64, 8192/16/4, 3)
// ---------------------------------------------------------------------------
__global__ void k_qkv_gemm(const float* __restrict__ x,
                           const __bf16* wqT, const __bf16* wkT, const __bf16* wvT,
                           const float* bq, const float* bk, const float* bv,
                           __bf16* q, __bf16* k, __bf16* v) {
    __shared__ __bf16 bst[2][64 * 32];      // double-buffered 4KB weight tiles

    int z = blockIdx.z;
    const __bf16* wT   = (z == 0) ? wqT : (z == 1) ? wkT : wvT;
    const float*  bias = (z == 0) ? bq  : (z == 1) ? bk  : bv;
    __bf16*       out  = (z == 0) ? q   : (z == 1) ? k   : v;

    int tid = threadIdx.x;
    int lane = tid & 31, wave = tid >> 5;
    int g = lane >> 4, hn = lane & 15;
    int colBase = blockIdx.x * 64;
    int rowBase = (blockIdx.y * 4 + wave) * 16;
    const __bf16* wtile = wT + (size_t)colBase * F_;

    v8f c[4];
#pragma unroll
    for (int t = 0; t < 4; ++t) c[t] = zero8();

    stage_tile(wtile, bst[0], 32, 64, F_, wave, tid);
    for (int kc = 0; kc < F_ / 32; ++kc) {
        int kk = kc * 32;
        bool more = (kc + 1) < F_ / 32;
        if (more) stage_tile(wtile + kk + 32, bst[(kc + 1) & 1], 32, 64, F_, wave, tid);
        finish_stage(more, wave);
        __syncthreads();
        v16bf a = load_a_f32(x, F_, rowBase + hn, kk, g);
        const __bf16* bb = bst[kc & 1];
#pragma unroll
        for (int t = 0; t < 4; ++t) {
            v16bf b = *(const v16bf*)(bb + (t * 16 + hn) * 32 + 16 * g);
            c[t] = wmma_bf16(a, b, c[t]);
        }
        __syncthreads();
    }
#pragma unroll
    for (int t = 0; t < 4; ++t) {
        int   col = colBase + t * 16 + hn;
        float bb  = bias[col];
#pragma unroll
        for (int v8 = 0; v8 < 8; ++v8) {
            int row = rowBase + v8 + 8 * g;
            out[(size_t)row * COLS_ + col] = (__bf16)(c[t][v8] + bb);
        }
    }
}

// ---------------------------------------------------------------------------
// Kernel 3: prep — k2[b,h,i] = sum_d k^2 ;  vT[b,h,d,i] = v[b,i,h,d]
// ---------------------------------------------------------------------------
__global__ void k_prep(const __bf16* __restrict__ kmat,
                       const __bf16* __restrict__ vmat,
                       float* __restrict__ k2, __bf16* __restrict__ vT) {
    int idx = blockIdx.x * 256 + threadIdx.x;
    int i  = idx & (S_ - 1);
    int bh = idx >> 11;
    int b  = bh >> 3;
    int h  = bh & 7;
    const __bf16* kp = kmat + ((size_t)(b * S_ + i) * COLS_ + h * DH_);
    const __bf16* vp = vmat + ((size_t)(b * S_ + i) * COLS_ + h * DH_);
    float s = 0.f;
#pragma unroll 8
    for (int d = 0; d < DH_; ++d) {
        float f = (float)kp[d];
        s += f * f;
    }
    k2[idx] = s;
#pragma unroll 8
    for (int d = 0; d < DH_; ++d)
        vT[((size_t)bh * DH_ + d) * S_ + i] = vp[d];
}

// ---------------------------------------------------------------------------
// Kernel 4: pass A — m[i], rinv[i].  Waves share bh: K tiles (32 j x 64 d)
// staged via TDM; q fragments resident per wave.
// grid = (B*H, S/16/4), blockDim=128
// ---------------------------------------------------------------------------
__global__ void k_softmax_stats(const __bf16* __restrict__ q,
                                const __bf16* __restrict__ kmat,
                                const float* __restrict__ k2,
                                float* __restrict__ m_out,
                                float* __restrict__ rinv_out) {
    __shared__ __bf16 kst[2][32 * 64];      // double-buffered 4KB K tiles

    int bh = blockIdx.x;
    int b = bh >> 3, h = bh & 7;
    int tid = threadIdx.x;
    int lane = tid & 31, wave = tid >> 5;
    int g = lane >> 4, hn = lane & 15;
    int iBase = (blockIdx.y * 4 + wave) * 16;

    const __bf16* qbase = q    + (size_t)b * S_ * COLS_ + h * DH_;
    const __bf16* kbase = kmat + (size_t)b * S_ * COLS_ + h * DH_;
    const float*  k2b   = k2   + (size_t)bh * S_;

    v16bf aq0 = load_a_bf16(qbase, COLS_, iBase + hn, 0,  g);
    v16bf aq1 = load_a_bf16(qbase, COLS_, iBase + hn, 32, g);

    float m[8], r[8];
#pragma unroll
    for (int v8 = 0; v8 < 8; ++v8) { m[v8] = -1e30f; r[v8] = 0.f; }

    stage_tile(kbase, kst[0], 64, 32, COLS_, wave, tid);
    for (int jc = 0; jc < S_ / 32; ++jc) {
        int j0 = jc * 32;
        bool more = (jc + 1) < S_ / 32;
        if (more)
            stage_tile(kbase + (size_t)(j0 + 32) * COLS_, kst[(jc + 1) & 1],
                       64, 32, COLS_, wave, tid);
        finish_stage(more, wave);
        __syncthreads();
        const __bf16* kt = kst[jc & 1];
#pragma unroll
        for (int t2 = 0; t2 < 2; ++t2) {
            v8f c = zero8();
            c = wmma_bf16(aq0, *(const v16bf*)(kt + (t2 * 16 + hn) * 64 + 16 * g), c);
            c = wmma_bf16(aq1, *(const v16bf*)(kt + (t2 * 16 + hn) * 64 + 32 + 16 * g), c);
            float k2j = k2b[j0 + t2 * 16 + hn];
#pragma unroll
            for (int v8 = 0; v8 < 8; ++v8) {
                float s  = 2.f * c[v8] - k2j;
                float mo = m[v8];
                float mn = fmaxf(mo, s);
                r[v8] = r[v8] * __expf(mo - mn) + __expf(s - mn);
                m[v8] = mn;
            }
        }
        __syncthreads();
    }
#pragma unroll
    for (int v8 = 0; v8 < 8; ++v8) {
#pragma unroll
        for (int off = 1; off < 16; off <<= 1) {
            float mo = __shfl_xor(m[v8], off, 16);
            float ro = __shfl_xor(r[v8], off, 16);
            float mn = fmaxf(m[v8], mo);
            r[v8] = r[v8] * __expf(m[v8] - mn) + ro * __expf(mo - mn);
            m[v8] = mn;
        }
    }
    if (hn == 0) {
#pragma unroll
        for (int v8 = 0; v8 < 8; ++v8) {
            int i = iBase + v8 + 8 * g;
            m_out[(size_t)bh * S_ + i]    = m[v8];
            rinv_out[(size_t)bh * S_ + i] = 1.f / r[v8];
        }
    }
}

// ---------------------------------------------------------------------------
// Kernel 5: pass B — attn[b,j,h,:] = sum_i p[i,j] v[i,:]  (P^T x V).
// V^T tiles (64 d x 32 i) staged via TDM (shared by all waves of the block);
// per-wave P^T transpose through LDS; 8 WMMAs per 32-i chunk.
// grid = (B*H, S/16/4), blockDim=128
// ---------------------------------------------------------------------------
__global__ void k_attn_accum(const __bf16* __restrict__ q,
                             const __bf16* __restrict__ kmat,
                             const __bf16* __restrict__ vT,
                             const float* __restrict__ k2,
                             const float* __restrict__ m_in,
                             const float* __restrict__ rinv_in,
                             __bf16* __restrict__ attn) {
    __shared__ __bf16 pst[4][16 * 32];      // per-wave P^T tiles (4KB)
    __shared__ __bf16 vst[2][64 * 32];      // double-buffered V^T tiles (8KB)

    int bh = blockIdx.x;
    int b = bh >> 3, h = bh & 7;
    int tid = threadIdx.x;
    int lane = tid & 31, wave = tid >> 5;
    int g = lane >> 4, hn = lane & 15;
    int jBase = (blockIdx.y * 4 + wave) * 16;

    const __bf16* qbase = q    + (size_t)b * S_ * COLS_ + h * DH_;
    const __bf16* kbase = kmat + (size_t)b * S_ * COLS_ + h * DH_;
    const __bf16* vtb   = vT   + (size_t)bh * DH_ * S_;

    v16bf bk0 = load_bt(kbase, COLS_, jBase + hn, 0,  g);
    v16bf bk1 = load_bt(kbase, COLS_, jBase + hn, 32, g);
    float k2j = k2[(size_t)bh * S_ + jBase + hn];

    v8f acc[4];
#pragma unroll
    for (int t = 0; t < 4; ++t) acc[t] = zero8();

    __bf16* myl = &pst[wave][0];

    stage_tile(vtb, vst[0], 32, 64, S_, wave, tid);
    for (int ic = 0; ic < S_ / 32; ++ic) {
        int i0 = ic * 32;
        bool more = (ic + 1) < S_ / 32;
        if (more) stage_tile(vtb + i0 + 32, vst[(ic + 1) & 1], 32, 64, S_, wave, tid);
        finish_stage(more, wave);
        __syncthreads();

        float mAll = m_in[(size_t)bh * S_ + i0 + lane];
        float rAll = rinv_in[(size_t)bh * S_ + i0 + lane];

#pragma unroll
        for (int t = 0; t < 2; ++t) {
            int ib = i0 + t * 16;
            v16bf aq0 = load_a_bf16(qbase, COLS_, ib + hn, 0,  g);
            v16bf aq1 = load_a_bf16(qbase, COLS_, ib + hn, 32, g);
            v8f c = zero8();
            c = wmma_bf16(aq0, bk0, c);
            c = wmma_bf16(aq1, bk1, c);
            v8bf pk;
#pragma unroll
            for (int v8 = 0; v8 < 8; ++v8) {
                int   src = t * 16 + v8 + 8 * g;
                float mi  = __shfl(mAll, src, 32);
                float ri  = __shfl(rAll, src, 32);
                float s   = 2.f * c[v8] - k2j;
                pk[v8] = (__bf16)(__expf(s - mi) * ri);
            }
            *(v8bf*)(myl + hn * 32 + t * 16 + 8 * g) = pk;
        }
        asm volatile("s_wait_dscnt 0" ::: "memory");

        v16bf ap = load_a_bf16(myl, 32, hn, 0, g);
        const __bf16* vt = vst[ic & 1];
#pragma unroll
        for (int t = 0; t < 4; ++t) {
            v16bf bv = *(const v16bf*)(vt + (t * 16 + hn) * 32 + 16 * g);
            acc[t] = wmma_bf16(ap, bv, acc[t]);
        }
        __syncthreads();
    }
#pragma unroll
    for (int t = 0; t < 4; ++t) {
        int d = t * 16 + hn;
#pragma unroll
        for (int v8 = 0; v8 < 8; ++v8) {
            int j = jBase + v8 + 8 * g;
            attn[(size_t)(b * S_ + j) * COLS_ + h * DH_ + d] = (__bf16)acc[t][v8];
        }
    }
}

// ---------------------------------------------------------------------------
// Kernel 6: output GEMM + bias + LeakyReLU, TDM-staged Wo panels.
// grid = (512/64, 8192/16/4), blockDim=128
// ---------------------------------------------------------------------------
__global__ void k_out_gemm(const __bf16* __restrict__ attn,
                           const __bf16* __restrict__ woT,
                           const float* __restrict__ bo,
                           float* __restrict__ out) {
    __shared__ __bf16 bst[2][64 * 32];

    int tid = threadIdx.x;
    int lane = tid & 31, wave = tid >> 5;
    int g = lane >> 4, hn = lane & 15;
    int colBase = blockIdx.x * 64;
    int rowBase = (blockIdx.y * 4 + wave) * 16;
    const __bf16* wtile = woT + (size_t)colBase * COLS_;

    v8f c[4];
#pragma unroll
    for (int t = 0; t < 4; ++t) c[t] = zero8();

    stage_tile(wtile, bst[0], 32, 64, COLS_, wave, tid);
    for (int kc = 0; kc < COLS_ / 32; ++kc) {
        int kk = kc * 32;
        bool more = (kc + 1) < COLS_ / 32;
        if (more) stage_tile(wtile + kk + 32, bst[(kc + 1) & 1], 32, 64, COLS_, wave, tid);
        finish_stage(more, wave);
        __syncthreads();
        v16bf a = load_a_bf16(attn, COLS_, rowBase + hn, kk, g);
        const __bf16* bb = bst[kc & 1];
#pragma unroll
        for (int t = 0; t < 4; ++t) {
            v16bf b = *(const v16bf*)(bb + (t * 16 + hn) * 32 + 16 * g);
            c[t] = wmma_bf16(a, b, c[t]);
        }
        __syncthreads();
    }
#pragma unroll
    for (int t = 0; t < 4; ++t) {
        int   col = colBase + t * 16 + hn;
        float bb  = bo[col];
#pragma unroll
        for (int v8 = 0; v8 < 8; ++v8) {
            int   row = rowBase + v8 + 8 * g;
            float val = c[t][v8] + bb;
            val = (val >= 0.f) ? val : NEG_SLOPE * val;
            out[(size_t)row * OUT_ + col] = val;
        }
    }
}

// ---------------------------------------------------------------------------
// Host launcher
// ---------------------------------------------------------------------------
extern "C" void kernel_launch(void* const* d_in, const int* in_sizes, int n_in,
                              void* d_out, int out_size, void* d_ws, size_t ws_size,
                              hipStream_t stream) {
    const float* x  = (const float*)d_in[0];
    const float* Wq = (const float*)d_in[1];
    const float* bq = (const float*)d_in[2];
    const float* Wk = (const float*)d_in[3];
    const float* bk = (const float*)d_in[4];
    const float* Wv = (const float*)d_in[5];
    const float* bv = (const float*)d_in[6];
    const float* Wo = (const float*)d_in[7];
    const float* bo = (const float*)d_in[8];
    float* out = (float*)d_out;

    char* ws = (char*)d_ws;
    size_t off = 0;
    auto alloc = [&](size_t bytes) {
        size_t o = off;
        off = (off + bytes + 255) & ~(size_t)255;
        return o;
    };
    const size_t wBytes    = (size_t)512 * 512 * 2;
    const size_t tokBytes  = (size_t)N_ * COLS_ * 2;
    const size_t statBytes = (size_t)B_ * H_ * S_ * 4;

    __bf16* wqT  = (__bf16*)(ws + alloc(wBytes));
    __bf16* wkT  = (__bf16*)(ws + alloc(wBytes));
    __bf16* wvT  = (__bf16*)(ws + alloc(wBytes));
    __bf16* woT  = (__bf16*)(ws + alloc(wBytes));
    __bf16* qb   = (__bf16*)(ws + alloc(tokBytes));
    __bf16* kb   = (__bf16*)(ws + alloc(tokBytes));
    __bf16* vb   = (__bf16*)(ws + alloc(tokBytes));
    __bf16* vT   = (__bf16*)(ws + alloc(tokBytes));
    float*  k2   = (float*)(ws + alloc(statBytes));
    float*  mbuf = (float*)(ws + alloc(statBytes));
    float*  rinv = (float*)(ws + alloc(statBytes));
    __bf16* attn = (__bf16*)(ws + alloc(tokBytes));
    (void)ws_size; (void)in_sizes; (void)n_in; (void)out_size;

    k_transpose_w<<<dim3(512 * 512 / 256, 1, 4), 256, 0, stream>>>(
        Wq, Wk, Wv, Wo, wqT, wkT, wvT, woT);

    k_qkv_gemm<<<dim3(COLS_ / 64, N_ / 16 / 4, 3), 128, 0, stream>>>(
        x, wqT, wkT, wvT, bq, bk, bv, qb, kb, vb);

    k_prep<<<dim3(B_ * H_ * S_ / 256), 256, 0, stream>>>(kb, vb, k2, vT);

    k_softmax_stats<<<dim3(B_ * H_, S_ / 16 / 4), 128, 0, stream>>>(
        qb, kb, k2, mbuf, rinv);

    k_attn_accum<<<dim3(B_ * H_, S_ / 16 / 4), 128, 0, stream>>>(
        qb, kb, vT, k2, mbuf, rinv, attn);

    k_out_gemm<<<dim3(OUT_ / 64, N_ / 16 / 4), 128, 0, stream>>>(
        attn, woT, bo, out);
}
